// HierarchicalBiasPredictionModel_13494787244225
// MI455X (gfx1250) — matrix operand order repaired and checked by reading, not compiled, not run on hardware
//
#include <hip/hip_runtime.h>

typedef __bf16 bf16_t;
typedef __bf16 v16bf __attribute__((ext_vector_type(16)));
typedef float  v8f   __attribute__((ext_vector_type(8)));

#define WMMA_BF16(a, b, c) \
  __builtin_amdgcn_wmma_f32_16x16x32_bf16(false, (a), false, (b), (short)0, (c), false, false)

// --------------------------------------------------------------------------
// CDNA5 async global->LDS staging (ASYNCcnt path), with safe fallback.
// Builtin signature (from hipcc diagnostic): param0 = v4i (vector_size 16) in
// global ("__device__") AS, param1 = LDS pointer, then imm offset + cpol.
// --------------------------------------------------------------------------
#if defined(__HIP_DEVICE_COMPILE__) && __has_builtin(__builtin_amdgcn_global_load_async_to_lds_b128)
#define HAVE_ASYNC_LDS 1
typedef int v4i_t __attribute__((vector_size(16)));
typedef v4i_t __attribute__((address_space(1)))* g128_t;
typedef v4i_t __attribute__((address_space(3)))* l128_t;
__device__ __forceinline__ void async_copy_b128(const void* g, void* l) {
  __builtin_amdgcn_global_load_async_to_lds_b128((g128_t)g, (l128_t)l, 0, 0);
}
#else
#define HAVE_ASYNC_LDS 0
#endif

__device__ __forceinline__ void wait_async0() {
#if defined(__HIP_DEVICE_COMPILE__)
#if __has_builtin(__builtin_amdgcn_s_wait_asynccnt)
  __builtin_amdgcn_s_wait_asynccnt(0);
#else
  asm volatile("s_wait_asynccnt 0x0" ::: "memory");
#endif
#endif
}

// --------------------------------------------------------------------------
// Fragment loaders (contiguous k order, wave32)
// A (16x32 bf16): lane m = lane&15, kg = lane>>4; elems 0..15 -> K=kg*16+e.
// B (32x16 bf16): lane n = lane&15; lanes 0-15: K=e, lanes 16-31: K=16+e.
// --------------------------------------------------------------------------
__device__ __forceinline__ v16bf load_a_frag(const bf16_t* row_k0, int lg) {
  v16bf a;
  const bf16_t* p = row_k0 + lg * 16;
#pragma unroll
  for (int e = 0; e < 8; ++e) a[e] = p[e];
#pragma unroll
  for (int e = 0; e < 8; ++e) a[8 + e] = p[8 + e];
  return a;
}

__device__ __forceinline__ v16bf load_b_frag_contig(const bf16_t* p) {
  v16bf b;
#pragma unroll
  for (int e = 0; e < 16; ++e) b[e] = p[e];
  return b;
}

__device__ __forceinline__ v16bf load_b_frag_strided(const bf16_t* p, int stride) {
  v16bf b;
#pragma unroll
  for (int e = 0; e < 16; ++e) b[e] = p[e * stride];
  return b;
}

// --------------------------------------------------------------------------
// fp32 -> bf16 conversion (n % 4 == 0)
// --------------------------------------------------------------------------
__global__ __launch_bounds__(256)
void f32_to_bf16_k(const float* __restrict__ in, bf16_t* __restrict__ out, long n) {
  long i = ((long)blockIdx.x * 256 + threadIdx.x) * 4;
  if (i < n) {
    float4 f = *reinterpret_cast<const float4*>(in + i);
    out[i + 0] = (bf16_t)f.x;
    out[i + 1] = (bf16_t)f.y;
    out[i + 2] = (bf16_t)f.z;
    out[i + 3] = (bf16_t)f.w;
  }
}

// --------------------------------------------------------------------------
// GEMM: Out[M,N](bf16) = A[M,K](bf16) @ W[N,K](bf16)^T + bias[N](f32)
// N % 128 == 0, K % 32 == 0.  256 threads = 8 waves; 128x128 tile;
// wave grid 4(M) x 2(N); 2x4 16x16 fragments per wave.
// Double-buffered LDS; async global->LDS copies on the CDNA5 path.
// GUARD_M=false: M % 128 == 0 (branch-free staging & stores).
// --------------------------------------------------------------------------
template <bool GUARD_M>
__global__ __launch_bounds__(256)
void gemm_wmma_bf16(const bf16_t* __restrict__ A, const bf16_t* __restrict__ W,
                    const float* __restrict__ bias, bf16_t* __restrict__ Out,
                    int M, int N, int K) {
  constexpr int BM = 128, BN = 128, BK = 32;
  __shared__ bf16_t As[2][BM * BK];
  __shared__ bf16_t Bs[2][BN * BK];

  const int tid = threadIdx.x;
  const int row0 = blockIdx.y * BM, col0 = blockIdx.x * BN;
  const int wave = tid >> 5, lane = tid & 31;
  const int lm = lane & 15, lg = lane >> 4;
  const int waveM = wave & 3, waveN = wave >> 2;
  const float4 fz4 = make_float4(0.f, 0.f, 0.f, 0.f);

  const v8f vz = {0.f, 0.f, 0.f, 0.f, 0.f, 0.f, 0.f, 0.f};
  v8f acc[2][4];
#pragma unroll
  for (int mi = 0; mi < 2; ++mi)
#pragma unroll
    for (int ni = 0; ni < 4; ++ni) acc[mi][ni] = vz;

  // Stage one 128x32 A tile + 128x32 W tile (16B chunks, 2 per thread each).
  auto stage = [&](int buf, int k0) {
#pragma unroll
    for (int p = 0; p < 2; ++p) {
      int q = p * 256 + tid;
      int r = q >> 2;            // tile row 0..127
      int cb = (q & 3) * 8;      // bf16 column chunk (16 bytes)
      int gr = row0 + r;
#if HAVE_ASYNC_LDS
      if (!GUARD_M || gr < M)
        async_copy_b128(A + (size_t)gr * K + k0 + cb, &As[buf][r * BK + cb]);
      else
        *reinterpret_cast<float4*>(&As[buf][r * BK + cb]) = fz4;
      async_copy_b128(W + (size_t)(col0 + r) * K + k0 + cb, &Bs[buf][r * BK + cb]);
#else
      float4 av = fz4;
      if (!GUARD_M || gr < M)
        av = *reinterpret_cast<const float4*>(A + (size_t)gr * K + k0 + cb);
      *reinterpret_cast<float4*>(&As[buf][r * BK + cb]) = av;
      float4 wv = *reinterpret_cast<const float4*>(W + (size_t)(col0 + r) * K + k0 + cb);
      *reinterpret_cast<float4*>(&Bs[buf][r * BK + cb]) = wv;
#endif
    }
  };

  const int nk = K / BK;
  stage(0, 0);
  for (int kt = 0; kt < nk; ++kt) {
    const int cur = kt & 1;
#if HAVE_ASYNC_LDS
    wait_async0();
#endif
    __syncthreads();
    if (kt + 1 < nk) stage(cur ^ 1, (kt + 1) * BK);

    v16bf af[2], bfr[4];
#pragma unroll
    for (int mi = 0; mi < 2; ++mi) {
      int m = waveM * 32 + mi * 16 + lm;
      af[mi] = load_a_frag(&As[cur][m * BK], lg);
    }
#pragma unroll
    for (int ni = 0; ni < 4; ++ni) {
      int n = waveN * 64 + ni * 16 + lm;
      bfr[ni] = load_b_frag_contig(&Bs[cur][n * BK + lg * 16]);
    }
#pragma unroll
    for (int mi = 0; mi < 2; ++mi)
#pragma unroll
      for (int ni = 0; ni < 4; ++ni)
        acc[mi][ni] = WMMA_BF16(af[mi], bfr[ni], acc[mi][ni]);
  }

  // Store C (16x16 f32 layout: lane col = lm; row = r + 8*lg) + bias, as bf16
#pragma unroll
  for (int mi = 0; mi < 2; ++mi)
#pragma unroll
    for (int ni = 0; ni < 4; ++ni) {
      int gn = col0 + waveN * 64 + ni * 16 + lm;
      float bv = bias ? bias[gn] : 0.f;
#pragma unroll
      for (int r = 0; r < 8; ++r) {
        int gm = row0 + waveM * 32 + mi * 16 + r + 8 * lg;
        if (!GUARD_M || gm < M)
          Out[(size_t)gm * N + gn] = (bf16_t)(acc[mi][ni][r] + bv);
      }
    }
}

// --------------------------------------------------------------------------
// Fused attention per (sequence, head): S = QK^T*scale (key-padding masked),
// P = softmax(S), O = P V.   d = 192 (E=768, H=4).  L tokens, Lpad in {16,64}.
// qkv: [NS*L, 2304] bf16 (q|k|v blocks of 768, head h at col h*192).
// out: [NS*L, 768] bf16.
// --------------------------------------------------------------------------
__global__ __launch_bounds__(128)
void attn_wmma(const bf16_t* __restrict__ qkv, const float* __restrict__ maskf,
               bf16_t* __restrict__ outp, int L, int Lpad, float scale) {
  const int seq = blockIdx.x, h = blockIdx.y;
  __shared__ bf16_t Qb[64 * 192];
  __shared__ bf16_t Kb[64 * 192];
  __shared__ bf16_t Vb[64 * 192];
  __shared__ float  Sb[64 * 64];
  __shared__ float  mcol[64];

  const int tid = threadIdx.x;
  const int wave = tid >> 5, lane = tid & 31;
  const int lm = lane & 15, lg = lane >> 4;
  const int d = 192, E3 = 2304;
  const float4 fz4 = make_float4(0.f, 0.f, 0.f, 0.f);

  // Load Q/K/V for this head in 16B chunks (zero-fill rows >= L)
  for (int idx = tid; idx < 64 * 24; idx += 128) {
    int t = idx / 24, cb = (idx % 24) * 8;
    float4 q4 = fz4, k4 = fz4, v4 = fz4;
    if (t < L) {
      size_t rowbase = (size_t)(seq * L + t) * E3 + h * 192 + cb;
      q4 = *reinterpret_cast<const float4*>(qkv + rowbase);
      k4 = *reinterpret_cast<const float4*>(qkv + rowbase + 768);
      v4 = *reinterpret_cast<const float4*>(qkv + rowbase + 1536);
    }
    *reinterpret_cast<float4*>(&Qb[t * 192 + cb]) = q4;
    *reinterpret_cast<float4*>(&Kb[t * 192 + cb]) = k4;
    *reinterpret_cast<float4*>(&Vb[t * 192 + cb]) = v4;
  }
  for (int idx = tid; idx < 64; idx += 128)
    mcol[idx] = (idx < L) ? maskf[seq * L + idx] : 0.f;
  __syncthreads();

  // Phase 1: S = Q K^T
  const v8f vz = {0.f, 0.f, 0.f, 0.f, 0.f, 0.f, 0.f, 0.f};
  const int nT = Lpad / 16;
  for (int tt = wave; tt < nT * nT; tt += 4) {
    int tm = tt / nT, tn = tt % nT;
    v8f acc = vz;
    for (int k0 = 0; k0 < d; k0 += 32) {
      v16bf a = load_a_frag(&Qb[(tm * 16 + lm) * d + k0], lg);
      // B[k][n] = K[n][k]: contiguous along k in row-major Kb
      v16bf b = load_b_frag_contig(&Kb[(tn * 16 + lm) * d + k0 + lg * 16]);
      acc = WMMA_BF16(a, b, acc);
    }
    int n = tn * 16 + lm;
    float mk = mcol[n];
#pragma unroll
    for (int r = 0; r < 8; ++r) {
      int m = tm * 16 + r + 8 * lg;
      float s = acc[r] * scale;
      Sb[m * 64 + n] = (mk > 0.f) ? s : -1e9f;
    }
  }
  __syncthreads();

  // P (bf16, [64][64]) aliases Qb; zero so K-dim padding to 32 is valid.
  bf16_t* P = Qb;
  for (int idx = tid; idx < 64 * 64; idx += 128) P[idx] = (bf16_t)0.f;
  __syncthreads();

  // Softmax per row (numerically safe even for fully-masked rows)
  if (tid < Lpad) {
    int m = tid;
    float mx = -3.0e38f;
    for (int n = 0; n < Lpad; ++n) mx = fmaxf(mx, Sb[m * 64 + n]);
    float sum = 0.f;
    for (int n = 0; n < Lpad; ++n) sum += __expf(Sb[m * 64 + n] - mx);
    float inv = 1.f / sum;
    for (int n = 0; n < Lpad; ++n)
      P[m * 64 + n] = (bf16_t)(__expf(Sb[m * 64 + n] - mx) * inv);
  }
  __syncthreads();

  // Phase 2: O = P @ V  (M=Lpad, N=192, K=max(Lpad,32))
  const int Kp = (Lpad < 32) ? 32 : Lpad;
  const int nT2 = (Lpad / 16) * 12;
  for (int tt = wave; tt < nT2; tt += 4) {
    int tm = tt / 12, tn = tt % 12;
    v8f acc = vz;
    for (int k0 = 0; k0 < Kp; k0 += 32) {
      v16bf a = load_a_frag(&P[(tm * 16 + lm) * 64 + k0], lg);
      // B[k][n] = V[k][n]: stride-192 along k
      v16bf b = load_b_frag_strided(&Vb[(size_t)(k0 + lg * 16) * d + tn * 16 + lm], d);
      acc = WMMA_BF16(a, b, acc);
    }
    int gn = tn * 16 + lm;
#pragma unroll
    for (int r = 0; r < 8; ++r) {
      int m = tm * 16 + r + 8 * lg;
      if (m < L)
        outp[(size_t)(seq * L + m) * 768 + h * 192 + gn] = (bf16_t)acc[r];
    }
  }
}

// --------------------------------------------------------------------------
// Masked mean over L: out[ns,e] = sum_l X[ns,l,e]*m[ns,l] / (sum_l m + 1e-10)
// --------------------------------------------------------------------------
__global__ __launch_bounds__(256)
void masked_mean_k(const bf16_t* __restrict__ X, const float* __restrict__ maskf,
                   bf16_t* __restrict__ outb, float* __restrict__ outf,
                   int L, int E) {
  int ns = blockIdx.x;
  int e = blockIdx.y * 256 + threadIdx.x;
  if (e >= E) return;
  float s = 0.f, msum = 0.f;
  for (int l = 0; l < L; ++l) {
    float mk = maskf[ns * L + l];
    msum += mk;
    s += mk * (float)X[((size_t)ns * L + l) * E + e];
  }
  float r = s / (msum + 1e-10f);
  if (outb) outb[(size_t)ns * E + e] = (bf16_t)r;
  if (outf) outf[(size_t)ns * E + e] = r;
}

// --------------------------------------------------------------------------
// Mask prep
// --------------------------------------------------------------------------
__global__ void mask_word_k(const int* __restrict__ am, float* __restrict__ mw, int n) {
  int i = blockIdx.x * 256 + threadIdx.x;
  if (i < n) mw[i] = (float)am[i];
}
__global__ void mask_sent_k(const int* __restrict__ am, float* __restrict__ ms) {
  int i = blockIdx.x * 256 + threadIdx.x;
  if (i < 1024) {
    int s = 0;
    for (int w = 0; w < 64; ++w) s += am[i * 64 + w];
    ms[i] = (s > 0) ? 1.f : 0.f;
  }
}
__global__ void mask_sec_k(const int* __restrict__ am, float* __restrict__ msec) {
  int i = threadIdx.x;
  if (i < 64) {
    int s = 0;
    for (int j = 0; j < 1024; ++j) s += am[i * 1024 + j];
    msec[i] = (s > 0) ? 1.f : 0.f;
  }
}

// --------------------------------------------------------------------------
// Tail: bp/hp projections, sigmoid gate, fuse, LN(256), fc1+relu, LN(128), fc2
// One block per batch row b; 256 threads.
// --------------------------------------------------------------------------
__device__ __forceinline__ float block_sum_256(float v, float* red, int tid) {
  red[tid] = v;
  __syncthreads();
  for (int s = 128; s > 0; s >>= 1) {
    if (tid < s) red[tid] += red[tid + s];
    __syncthreads();
  }
  float r = red[0];
  __syncthreads();
  return r;
}

__global__ __launch_bounds__(256)
void tail_k(const float* __restrict__ doc, const float* __restrict__ hand,
            const float* __restrict__ bp_w, const float* __restrict__ bp_b,
            const float* __restrict__ hp_w, const float* __restrict__ hp_b,
            const float* __restrict__ gate_w, const float* __restrict__ gate_b,
            const float* __restrict__ lng, const float* __restrict__ lnb,
            const float* __restrict__ fc1_w, const float* __restrict__ fc1_b,
            const float* __restrict__ ln1g, const float* __restrict__ ln1b,
            const float* __restrict__ fc2_w, const float* __restrict__ fc2_b,
            float* __restrict__ out) {
  const int b = blockIdx.x, tid = threadIdx.x;
  __shared__ float drow[768];
  __shared__ float bp[256], hp[256], fused[256], h1[128];
  __shared__ float red[256];

  for (int i = tid; i < 768; i += 256) drow[i] = doc[b * 768 + i];
  __syncthreads();

  {  // bert_proj and hand_proj
    float s = bp_b[tid];
    for (int k = 0; k < 768; ++k) s += drow[k] * bp_w[tid * 768 + k];
    bp[tid] = s;
    float t = hp_b[tid];
    for (int k = 0; k < 20; ++k) t += hand[b * 20 + k] * hp_w[tid * 20 + k];
    hp[tid] = t;
  }
  __syncthreads();

  {  // gate + fuse
    float s = gate_b[tid];
    for (int k = 0; k < 256; ++k) s += bp[k] * gate_w[tid * 512 + k];
    for (int k = 0; k < 256; ++k) s += hp[k] * gate_w[tid * 512 + 256 + k];
    float g = 1.f / (1.f + __expf(-s));
    fused[tid] = g * bp[tid] + (1.f - g) * hp[tid];
  }
  __syncthreads();

  {  // LayerNorm over 256
    float x = fused[tid];
    float mean = block_sum_256(x, red, tid) * (1.f / 256.f);
    float dv = x - mean;
    float var = block_sum_256(dv * dv, red, tid) * (1.f / 256.f);
    fused[tid] = dv * rsqrtf(var + 1e-5f) * lng[tid] + lnb[tid];
  }
  __syncthreads();

  if (tid < 128) {  // fc1 + relu
    float s = fc1_b[tid];
    for (int k = 0; k < 256; ++k) s += fused[k] * fc1_w[tid * 256 + k];
    h1[tid] = fmaxf(s, 0.f);
  }
  __syncthreads();

  {  // LayerNorm over 128
    float hv = (tid < 128) ? h1[tid] : 0.f;
    float mean = block_sum_256(hv, red, tid) * (1.f / 128.f);
    float dv = (tid < 128) ? (hv - mean) : 0.f;
    float var = block_sum_256(dv * dv, red, tid) * (1.f / 128.f);
    if (tid < 128) h1[tid] = dv * rsqrtf(var + 1e-5f) * ln1g[tid] + ln1b[tid];
  }
  __syncthreads();

  if (tid < 3) {  // fc2
    float s = fc2_b[tid];
    for (int k = 0; k < 128; ++k) s += h1[k] * fc2_w[tid * 128 + k];
    out[b * 3 + tid] = s;
  }
}

// --------------------------------------------------------------------------
// Orchestration
// --------------------------------------------------------------------------
extern "C" void kernel_launch(void* const* d_in, const int* in_sizes, int n_in,
                              void* d_out, int out_size, void* d_ws, size_t ws_size,
                              hipStream_t stream) {
  (void)in_sizes; (void)n_in; (void)out_size; (void)ws_size;
  const float* wemb   = (const float*)d_in[0];
  const int*   amask  = (const int*)d_in[1];
  const float* hand   = (const float*)d_in[2];
  const float* w_w_in = (const float*)d_in[3];
  const float* w_b_in = (const float*)d_in[4];
  const float* w_w_out= (const float*)d_in[5];
  const float* w_b_out= (const float*)d_in[6];
  const float* s_w_in = (const float*)d_in[7];
  const float* s_b_in = (const float*)d_in[8];
  const float* s_w_out= (const float*)d_in[9];
  const float* s_b_out= (const float*)d_in[10];
  const float* c_w_in = (const float*)d_in[11];
  const float* c_b_in = (const float*)d_in[12];
  const float* c_w_out= (const float*)d_in[13];
  const float* c_b_out= (const float*)d_in[14];
  const float* bp_w   = (const float*)d_in[15];
  const float* bp_b   = (const float*)d_in[16];
  const float* hp_w   = (const float*)d_in[17];
  const float* hp_b   = (const float*)d_in[18];
  // d_in[19..22] = fa_* cross-attention: output unused in reference -> skipped
  const float* gate_w = (const float*)d_in[23];
  const float* gate_b = (const float*)d_in[24];
  const float* fln_g  = (const float*)d_in[25];
  const float* fln_b  = (const float*)d_in[26];
  const float* fc1_w  = (const float*)d_in[27];
  const float* fc1_b  = (const float*)d_in[28];
  const float* ln1_g  = (const float*)d_in[29];
  const float* ln1_b  = (const float*)d_in[30];
  const float* fc2_w  = (const float*)d_in[31];
  const float* fc2_b  = (const float*)d_in[32];
  float* out = (float*)d_out;

  const int E = 768, E3 = 2304, H = 4;
  const int Mw = 65536;          // B*S*T*W tokens
  const int NSw = 1024, Lw = 64; // word-level sequences

  char* ws = (char*)d_ws;
  size_t off = 0;
  auto take = [&](size_t bytes) -> char* {
    char* p = ws + off;
    off = (off + bytes + 255) & ~(size_t)255;
    return p;
  };
  // bf16 copies of fp32 operands (feed all WMMA GEMMs)
  bf16_t* emb_bf  = (bf16_t*)take((size_t)Mw * E * 2);
  bf16_t* wwin_bf = (bf16_t*)take((size_t)E3 * E * 2);
  bf16_t* wwout_bf= (bf16_t*)take((size_t)E * E * 2);
  bf16_t* swin_bf = (bf16_t*)take((size_t)E3 * E * 2);
  bf16_t* swout_bf= (bf16_t*)take((size_t)E * E * 2);
  bf16_t* cwin_bf = (bf16_t*)take((size_t)E3 * E * 2);
  bf16_t* cwout_bf= (bf16_t*)take((size_t)E * E * 2);
  // pipeline buffers
  bf16_t* qkv_w = (bf16_t*)take((size_t)Mw * E3 * 2);
  bf16_t* aow   = (bf16_t*)take((size_t)Mw * E * 2);
  bf16_t* wo    = (bf16_t*)take((size_t)Mw * E * 2);
  bf16_t* sent  = (bf16_t*)take((size_t)1024 * E * 2);
  bf16_t* qkv_s = (bf16_t*)take((size_t)1024 * E3 * 2);
  bf16_t* aos   = (bf16_t*)take((size_t)1024 * E * 2);
  bf16_t* so    = (bf16_t*)take((size_t)1024 * E * 2);
  bf16_t* sec   = (bf16_t*)take((size_t)64 * E * 2);
  bf16_t* qkv_c = (bf16_t*)take((size_t)64 * E3 * 2);
  bf16_t* aoc   = (bf16_t*)take((size_t)64 * E * 2);
  bf16_t* doo   = (bf16_t*)take((size_t)64 * E * 2);
  float*  doc   = (float*)take((size_t)8 * E * 4);
  float*  mw    = (float*)take((size_t)65536 * 4);
  float*  ms    = (float*)take((size_t)1024 * 4);
  float*  msec  = (float*)take((size_t)64 * 4);

  const float scale = 0.07216878364870323f;  // 1/sqrt(192)

  auto cvt = [&](const float* src, bf16_t* dst, long n) {
    f32_to_bf16_k<<<dim3((unsigned)((n / 4 + 255) / 256)), 256, 0, stream>>>(src, dst, n);
  };
  cvt(wemb,    emb_bf,   (long)Mw * E);
  cvt(w_w_in,  wwin_bf,  (long)E3 * E);
  cvt(w_w_out, wwout_bf, (long)E * E);
  cvt(s_w_in,  swin_bf,  (long)E3 * E);
  cvt(s_w_out, swout_bf, (long)E * E);
  cvt(c_w_in,  cwin_bf,  (long)E3 * E);
  cvt(c_w_out, cwout_bf, (long)E * E);

  // Masks
  mask_word_k<<<dim3(256), dim3(256), 0, stream>>>(amask, mw, 65536);
  mask_sent_k<<<dim3(4), dim3(256), 0, stream>>>(amask, ms);
  mask_sec_k<<<dim3(1), dim3(64), 0, stream>>>(amask, msec);

  // ---- word level (M % 128 == 0: unguarded GEMM) ----
  gemm_wmma_bf16<false><<<dim3(E3 / 128, Mw / 128), 256, 0, stream>>>(
      emb_bf, wwin_bf, w_b_in, qkv_w, Mw, E3, E);
  attn_wmma<<<dim3(NSw, H), 128, 0, stream>>>(qkv_w, mw, aow, Lw, 64, scale);
  gemm_wmma_bf16<false><<<dim3(E / 128, Mw / 128), 256, 0, stream>>>(
      aow, wwout_bf, w_b_out, wo, Mw, E, E);
  masked_mean_k<<<dim3(NSw, 3), 256, 0, stream>>>(wo, mw, sent, nullptr, Lw, E);

  // ---- sentence level (64 sequences of L=16; M=1024 unguarded) ----
  gemm_wmma_bf16<false><<<dim3(E3 / 128, 8), 256, 0, stream>>>(
      sent, swin_bf, s_b_in, qkv_s, 1024, E3, E);
  attn_wmma<<<dim3(64, H), 128, 0, stream>>>(qkv_s, ms, aos, 16, 16, scale);
  gemm_wmma_bf16<false><<<dim3(E / 128, 8), 256, 0, stream>>>(
      aos, swout_bf, s_b_out, so, 1024, E, E);
  masked_mean_k<<<dim3(64, 3), 256, 0, stream>>>(so, ms, sec, nullptr, 16, E);

  // ---- section level (8 sequences of L=8, padded to 16; M=64 guarded) ----
  gemm_wmma_bf16<true><<<dim3(E3 / 128, 1), 256, 0, stream>>>(
      sec, cwin_bf, c_b_in, qkv_c, 64, E3, E);
  attn_wmma<<<dim3(8, H), 128, 0, stream>>>(qkv_c, msec, aoc, 8, 16, scale);
  gemm_wmma_bf16<true><<<dim3(E / 128, 1), 256, 0, stream>>>(
      aoc, cwout_bf, c_b_out, doo, 64, E, E);
  masked_mean_k<<<dim3(8, 3), 256, 0, stream>>>(doo, msec, nullptr, doc, 8, E);

  // ---- tail ----
  tail_k<<<dim3(8), 256, 0, stream>>>(doc, hand, bp_w, bp_b, hp_w, hp_b,
                                      gate_w, gate_b, fln_g, fln_b,
                                      fc1_w, fc1_b, ln1_g, ln1_b,
                                      fc2_w, fc2_b, out);
}
